// CrossAttention_94489280562
// MI455X (gfx1250) — matrix operand order repaired and checked by reading, not compiled, not run on hardware
//
#include <hip/hip_runtime.h>

// ---------------------------------------------------------------------------
// CrossAttention for MI455X (gfx1250): f16 WMMA everywhere, f32 accumulate.
// B=4, N=M=4096, C=512.
// Round 3: software-pipelined fragment loads (rotate-by-one buffers) so LDS
//          ds_load_b128 latency hides under v_wmma instead of s_wait_dscnt 0x0
//          serialization; shared dual-chain gemm512 helper for scores + proj.
// ---------------------------------------------------------------------------

typedef _Float16 h8v  __attribute__((ext_vector_type(8)));
typedef _Float16 v16h __attribute__((ext_vector_type(16)));
typedef float    v8f  __attribute__((ext_vector_type(8)));

typedef unsigned int u32x4 __attribute__((ext_vector_type(4)));
typedef int          i32x4 __attribute__((ext_vector_type(4)));
typedef int          i32x8 __attribute__((ext_vector_type(8)));

#define CC   512
#define NTOK 4096
#define NBAT 4
#define MT   64        // keys per flash iteration

union HU { v16h v; h8v h[2]; };

static __device__ __forceinline__ v8f wmma16(v16h a, v16h b, v8f c) {
  // D = A(16x32) * B(32x16) + C(16x16), f16 in / f32 out
  return __builtin_amdgcn_wmma_f32_16x16x32_f16(false, a, false, b,
                                                (short)0, c, false, false);
}

// A-fragment (16x32, row-major source, K-contiguous rows).
// CDNA5 layout: lane L (half = L>>4) holds row M=L&15,
//   elements 0..7  -> K = kbase + 8*half + e
//   elements 8..15 -> K = kbase + 16 + 8*half + (e-8)
static __device__ __forceinline__ v16h loadFragA(const _Float16* p, int stride,
                                                 int rowbase, int kbase, int lane) {
  const int half = lane >> 4;
  const int r = rowbase + (lane & 15);
  const _Float16* q = p + (size_t)r * stride + kbase + 8 * half;
  HU u;
  u.h[0] = *(const h8v*)(q);
  u.h[1] = *(const h8v*)(q + 16);
  return u.v;
}

// B-fragment (32x16) from a K-major ("column major") source:
// source[(colbase + lane&15)*stride + k], lane holds K = 8*half+e (e<8), 16+8*half+(e-8) (e>=8).
static __device__ __forceinline__ v16h loadFragB(const _Float16* p, int stride,
                                                 int colbase, int kbase, int lane) {
  const int half = lane >> 4;
  const int c = colbase + (lane & 15);
  const _Float16* q = p + (size_t)c * stride + kbase + 8 * half;
  HU u;
  u.h[0] = *(const h8v*)(q);
  u.h[1] = *(const h8v*)(q + 16);
  return u.v;
}

// ---------------------------------------------------------------------------
// 16x16 tile, contraction over 512: dual accumulator chains + pipelined
// fragment loads (next pair issued before current WMMAs consume theirs).
// A: row-major [row][k] (stride CC); B: K-major [col][k] (stride CC).
// ---------------------------------------------------------------------------
static __device__ __forceinline__ v8f gemm512(const _Float16* A, int arow,
                                              const _Float16* B, int bcol,
                                              int lane) {
  v8f acc0 = {0.f,0.f,0.f,0.f,0.f,0.f,0.f,0.f};
  v8f acc1 = {0.f,0.f,0.f,0.f,0.f,0.f,0.f,0.f};
  v16h a0 = loadFragA(A, CC, arow, 0,  lane);
  v16h b0 = loadFragB(B, CC, bcol, 0,  lane);
  v16h a1 = loadFragA(A, CC, arow, 32, lane);
  v16h b1 = loadFragB(B, CC, bcol, 32, lane);
#pragma unroll
  for (int p = 0; p < 8; p++) {
    v16h na0 = a0, nb0 = b0, na1 = a1, nb1 = b1;
    if (p < 7) {
      na0 = loadFragA(A, CC, arow, (2 * p + 2) * 32, lane);
      nb0 = loadFragB(B, CC, bcol, (2 * p + 2) * 32, lane);
      na1 = loadFragA(A, CC, arow, (2 * p + 3) * 32, lane);
      nb1 = loadFragB(B, CC, bcol, (2 * p + 3) * 32, lane);
    }
    acc0 = wmma16(a0, b0, acc0);
    acc1 = wmma16(a1, b1, acc1);
    a0 = na0; b0 = nb0; a1 = na1; b1 = nb1;
  }
  return acc0 + acc1;
}

// ---------------------------------------------------------------------------
// Tensor Data Mover: 2D tile load Global -> LDS (D# per CDNA5 ISA).
// tile rows (dim1) x tile cols (dim0), 2-byte elements, row stride stride0.
// ---------------------------------------------------------------------------
static __device__ __forceinline__ void tdm_load_2d(unsigned lds_addr,
                                                   const _Float16* gptr,
                                                   unsigned tensor_d0,
                                                   unsigned tensor_d1,
                                                   unsigned tile_d0,
                                                   unsigned tile_d1,
                                                   unsigned stride0) {
  unsigned long long ga = (unsigned long long)(uintptr_t)gptr;
  u32x4 g0;
  g0[0] = 1u;                                   // count=1, user descriptor
  g0[1] = lds_addr;                             // lds_addr [63:32]
  g0[2] = (unsigned)ga;                         // global_addr [95:64]
  g0[3] = (unsigned)((ga >> 32) & 0x01FFFFFFull) | (2u << 30);  // addr hi | type=2
  i32x8 g1;
  g1[0] = (int)(1u << 16);                      // data_size=1 (2 bytes)
  g1[1] = (int)((tensor_d0 & 0xFFFFu) << 16);   // tensor_dim0[15:0] @ 63:48
  g1[2] = (int)(((tensor_d0 >> 16) & 0xFFFFu) | ((tensor_d1 & 0xFFFFu) << 16));
  g1[3] = (int)(((tensor_d1 >> 16) & 0xFFFFu) | ((tile_d0 & 0xFFFFu) << 16));
  g1[4] = (int)(tile_d1 & 0xFFFFu);             // tile_dim1; tile_dim2=0
  g1[5] = (int)stride0;                         // tensor_dim0_stride[31:0]
  g1[6] = 0;                                    // stride0 hi | dim1_stride lo
  g1[7] = 0;
  i32x4 z4 = {0, 0, 0, 0};
#if __clang_major__ >= 23
  i32x8 z8 = {0, 0, 0, 0, 0, 0, 0, 0};
  __builtin_amdgcn_tensor_load_to_lds(g0, g1, z4, z4, z8, 0);
#else
  __builtin_amdgcn_tensor_load_to_lds(g0, g1, z4, z4, 0);
#endif
}

// ---------------------------------------------------------------------------
// Kernel 1: transpose+convert weights: W[cin][cout] f32 -> Wt[cout][cin] f16
// ---------------------------------------------------------------------------
__global__ __launch_bounds__(256) void xattn_wt_kernel(const float* __restrict__ W,
                                                       _Float16* __restrict__ Wt) {
  int idx  = blockIdx.x * 256 + threadIdx.x;   // 0 .. 512*512-1
  int cout = idx & (CC - 1);
  int cin  = idx >> 9;
  Wt[(size_t)cout * CC + cin] = (_Float16)W[(size_t)cin * CC + cout];
}

// ---------------------------------------------------------------------------
// Kernel 2: projection GEMM.  X[16384][512] f32, Wt[cout][cin] f16, bias f32.
// Block tile: 32 tokens x 64 couts, 8 waves = 2 token-subtiles x 4 cout-subtiles.
// VMODE==0 : Y[token][cout] f16 (row major)  -> used for Q (scaled) and K
// VMODE==1 : Y stored transposed per batch: Vt[b][cout][m] f16 -> used for V
// ---------------------------------------------------------------------------
template <int VMODE>
__global__ __launch_bounds__(256) void xattn_proj_kernel(const float* __restrict__ X,
                                                         const _Float16* __restrict__ Wt,
                                                         const float* __restrict__ bias,
                                                         _Float16* __restrict__ Y,
                                                         float scale) {
  __shared__ _Float16 sX[32 * CC];   // 32 KB, token-major (cin contiguous)
  __shared__ _Float16 sW[64 * CC];   // 64 KB, cout-major (cin contiguous)

  const int t0 = blockIdx.x * 32;    // token base
  const int c0 = blockIdx.y * 64;    // cout base

  // Stage X tile (f32 -> f16): 32*512 = 16384 halfs, 4 per thread-step.
  for (int i = threadIdx.x; i < 4096; i += 256) {
    int tok = i >> 7;
    int c   = (i & 127) * 4;
    const float4 f = *(const float4*)(X + (size_t)(t0 + tok) * CC + c);
    _Float16* d = sX + tok * CC + c;
    d[0] = (_Float16)f.x; d[1] = (_Float16)f.y;
    d[2] = (_Float16)f.z; d[3] = (_Float16)f.w;
  }
  // Stage Wt tile (already f16): 64*512 = 32768 halfs, 8 per thread-step.
  for (int i = threadIdx.x; i < 4096; i += 256) {
    int r = i >> 6;
    int c = (i & 63) * 8;
    *(h8v*)(sW + r * CC + c) = *(const h8v*)(Wt + (size_t)(c0 + r) * CC + c);
  }
  __syncthreads();

  const int w    = threadIdx.x >> 5;
  const int lane = threadIdx.x & 31;
  const int half = lane >> 4;
  const int tg   = w >> 2;   // token subtile (0..1)
  const int cg   = w & 3;    // cout subtile (0..3)

  if (VMODE == 0) {
    // A = Wt rows (M = cout), B = X columns (N = token)
    v8f acc = gemm512(sW, cg * 16, sX, tg * 16, lane);
    const int tok = t0 + tg * 16 + (lane & 15);
    const int cb  = c0 + cg * 16 + 8 * half;   // 8 contiguous couts per lane
    h8v o;
#pragma unroll
    for (int v = 0; v < 8; v++)
      o[v] = (_Float16)((acc[v] + bias[cb + v]) * scale);
    *(h8v*)(Y + (size_t)tok * CC + cb) = o;
  } else {
    // A = X rows (M = token), B = Wt columns (N = cout)
    v8f acc = gemm512(sX, tg * 16, sW, cg * 16, lane);
    const int cout    = c0 + cg * 16 + (lane & 15);
    const float bb    = bias[cout];
    const int tokbase = t0 + tg * 16 + 8 * half;   // 8 contiguous tokens per lane
    const int b_      = tokbase >> 12;             // batch
    const int m       = tokbase & (NTOK - 1);
    h8v o;
#pragma unroll
    for (int v = 0; v < 8; v++)
      o[v] = (_Float16)(acc[v] + bb);
    *(h8v*)(Y + ((size_t)b_ * CC + cout) * NTOK + m) = o;
  }
}

// ---------------------------------------------------------------------------
// Kernel 3: fused flash attention.
// Block: 32 queries, stream 64 keys/iter over M=4096 (64 iters).
// K tile (64x512) and Vt tile (512x64) staged by the Tensor Data Mover.
// Scores: all 8 waves, one 16x16 sub-tile each (2 qg x 4 kg), pipelined dual
// accumulator chains.  PV: 2 qg x 4 cg x 8 c-subtiles, pipelined B-fragments.
// ---------------------------------------------------------------------------
__global__ __launch_bounds__(256) void xattn_flash_kernel(const _Float16* __restrict__ Q,
                                                          const _Float16* __restrict__ K,
                                                          const _Float16* __restrict__ Vt,
                                                          float* __restrict__ Out) {
  __shared__ _Float16 sK[MT * CC];    // 64 KB : keys row-major (c contiguous)
  __shared__ _Float16 sV[CC * MT];    // 64 KB : Vt tile, c-major (m contiguous)
  __shared__ float    sS[32 * MT];    //  8 KB : raw scores
  __shared__ _Float16 sP[32 * MT];    //  4 KB : exp probs (A-matrix for PV)
  __shared__ float    sMax[32], sSum[32], sAlpha[32];
  __shared__ float    sPart[32 * 8];

  const int b  = blockIdx.y;
  const int q0 = blockIdx.x * 32;

  const _Float16* Qb = Q  + ((size_t)b * NTOK + q0) * CC;
  const _Float16* Kb = K  + (size_t)b * NTOK * CC;
  const _Float16* Vb = Vt + (size_t)b * CC * NTOK;

  const int w    = threadIdx.x >> 5;
  const int lane = threadIdx.x & 31;
  const int half = lane >> 4;
  const int qg   = w >> 2;   // query group (0..1) for scores and PV
  const int cg   = w & 3;    // PV channel group (0..3), 128 channels each
  const int kg   = w & 3;    // score key group (0..3)

  if (threadIdx.x < 32) { sMax[threadIdx.x] = -1e30f; sSum[threadIdx.x] = 0.f; }

  v8f oacc[8];
#pragma unroll
  for (int s = 0; s < 8; s++) oacc[s] = (v8f){0.f,0.f,0.f,0.f,0.f,0.f,0.f,0.f};

  for (int it = 0; it < NTOK / MT; it++) {
    const int m0 = it * MT;

    // --- stage K + Vt tiles via Tensor Data Mover (wave 0 issues both) ---
    if (w == 0) {
      // K tile: rows = keys (64), cols = c (512), row stride 512
      tdm_load_2d((unsigned)(uintptr_t)sK, Kb + (size_t)m0 * CC,
                  CC, NTOK, CC, MT, CC);
      // Vt tile: rows = c (512), cols = m (64), row stride 4096
      tdm_load_2d((unsigned)(uintptr_t)sV, Vb + m0,
                  NTOK, CC, MT, CC, NTOK);
      __builtin_amdgcn_s_wait_tensorcnt(0);
    }
    __syncthreads();

    // --- scores: every wave computes one 16x16 sub-tile over full C ---
    {
      v8f s = gemm512(Qb, qg * 16, sK, kg * 16, lane);   // A: global/L2, B: LDS
#pragma unroll
      for (int v = 0; v < 8; v++) {
        int qq = qg * 16 + v + 8 * half;
        sS[qq * MT + kg * 16 + (lane & 15)] = s[v];
      }
    }
    __syncthreads();

    // --- online softmax: stage 1, per-thread partial max (8 thr / row) ---
    {
      int r = threadIdx.x >> 3, j0 = (threadIdx.x & 7) * 8;
      const float* row = sS + r * MT + j0;
      float mx = row[0];
#pragma unroll
      for (int j = 1; j < 8; j++) mx = fmaxf(mx, row[j]);
      sPart[r * 8 + (threadIdx.x & 7)] = mx;
    }
    __syncthreads();
    if (threadIdx.x < 32) {
      int r = threadIdx.x;
      float mx = sPart[r * 8];
#pragma unroll
      for (int k = 1; k < 8; k++) mx = fmaxf(mx, sPart[r * 8 + k]);
      float mOld  = sMax[r];
      float mNew  = fmaxf(mOld, mx);
      float alpha = __expf(mOld - mNew);
      sAlpha[r] = alpha;
      sMax[r]   = mNew;
      sSum[r]  *= alpha;
    }
    __syncthreads();
    // --- stage 2: exponentiate, write P (f16), partial row sums ---
    {
      int r = threadIdx.x >> 3, j0 = (threadIdx.x & 7) * 8;
      float mNew = sMax[r];
      float acc = 0.f;
#pragma unroll
      for (int j = 0; j < 8; j++) {
        float p = __expf(sS[r * MT + j0 + j] - mNew);
        acc += p;
        sP[r * MT + j0 + j] = (_Float16)p;
      }
      sPart[r * 8 + (threadIdx.x & 7)] = acc;
    }
    __syncthreads();
    if (threadIdx.x < 32) {
      int r = threadIdx.x;
      float s2 = 0.f;
#pragma unroll
      for (int k = 0; k < 8; k++) s2 += sPart[r * 8 + k];
      sSum[r] += s2;
    }

    // --- PV: rescale running O by alpha, accumulate P(16x64) x V(64x16),
    //     pipelined B-fragment loads (next c-subtile in flight) ---
    {
      float al[8];
#pragma unroll
      for (int v = 0; v < 8; v++) al[v] = sAlpha[qg * 16 + v + 8 * half];
#pragma unroll
      for (int sub = 0; sub < 8; sub++)
#pragma unroll
        for (int v = 0; v < 8; v++) oacc[sub][v] *= al[v];

      v16h a0 = loadFragA(sP, MT, qg * 16, 0,  lane);  // shared across c-subtiles
      v16h a1 = loadFragA(sP, MT, qg * 16, 32, lane);
      v16h bf0 = loadFragB(sV, MT, cg * 128, 0,  lane);
      v16h bf1 = loadFragB(sV, MT, cg * 128, 32, lane);
#pragma unroll
      for (int sub = 0; sub < 8; sub++) {
        v16h nb0 = bf0, nb1 = bf1;
        if (sub < 7) {
          nb0 = loadFragB(sV, MT, cg * 128 + (sub + 1) * 16, 0,  lane);
          nb1 = loadFragB(sV, MT, cg * 128 + (sub + 1) * 16, 32, lane);
        }
        oacc[sub] = wmma16(a0, bf0, oacc[sub]);
        oacc[sub] = wmma16(a1, bf1, oacc[sub]);
        bf0 = nb0; bf1 = nb1;
      }
    }
    __syncthreads();   // protect sK/sV/sS/sP/sPart before next staging
  }

  // --- epilogue: normalize by running sum, store f32 output ---
  float inv[8];
#pragma unroll
  for (int v = 0; v < 8; v++)
    inv[v] = 1.0f / sSum[qg * 16 + v + 8 * half];

#pragma unroll
  for (int sub = 0; sub < 8; sub++) {
#pragma unroll
    for (int v = 0; v < 8; v++) {
      int qq = qg * 16 + v + 8 * half;
      int c  = cg * 128 + sub * 16 + (lane & 15);
      Out[((size_t)b * NTOK + q0 + qq) * CC + c] = oacc[sub][v] * inv[v];
    }
  }
}

// ---------------------------------------------------------------------------
// Launcher
// ---------------------------------------------------------------------------
extern "C" void kernel_launch(void* const* d_in, const int* in_sizes, int n_in,
                              void* d_out, int out_size, void* d_ws, size_t ws_size,
                              hipStream_t stream) {
  const float* rgb   = (const float*)d_in[0];
  const float* depth = (const float*)d_in[1];
  const float* Wq    = (const float*)d_in[2];
  const float* bq    = (const float*)d_in[3];
  const float* Wk    = (const float*)d_in[4];
  const float* bk    = (const float*)d_in[5];
  const float* Wv    = (const float*)d_in[6];
  const float* bv    = (const float*)d_in[7];
  float* out = (float*)d_out;

  // Workspace layout (f16): 3 transposed weights + Q + K + Vt  (~49.5 MB)
  _Float16* Wtq = (_Float16*)d_ws;
  _Float16* Wtk = Wtq + (size_t)CC * CC;
  _Float16* Wtv = Wtk + (size_t)CC * CC;
  _Float16* Qf  = Wtv + (size_t)CC * CC;
  _Float16* Kf  = Qf + (size_t)NBAT * NTOK * CC;
  _Float16* Vtf = Kf + (size_t)NBAT * NTOK * CC;

  const float scaleQ = 0.044194173824159216f;  // 1/sqrt(512)

  // 1) transpose/convert weights
  xattn_wt_kernel<<<(CC * CC) / 256, 256, 0, stream>>>(Wq, Wtq);
  xattn_wt_kernel<<<(CC * CC) / 256, 256, 0, stream>>>(Wk, Wtk);
  xattn_wt_kernel<<<(CC * CC) / 256, 256, 0, stream>>>(Wv, Wtv);

  // 2) projections (Q has softmax scale folded in)
  dim3 pgrid((NBAT * NTOK) / 32, CC / 64);
  xattn_proj_kernel<0><<<pgrid, 256, 0, stream>>>(rgb,   Wtq, bq, Qf,  scaleQ);
  xattn_proj_kernel<0><<<pgrid, 256, 0, stream>>>(depth, Wtk, bk, Kf,  1.0f);
  xattn_proj_kernel<1><<<pgrid, 256, 0, stream>>>(depth, Wtv, bv, Vtf, 1.0f);

  // 3) fused flash attention
  dim3 agrid(NTOK / 32, NBAT);
  xattn_flash_kernel<<<agrid, 256, 0, stream>>>(Qf, Kf, Vtf, out);
}
